// AttentionModel_65859028517128
// MI455X (gfx1250) — matrix operand order repaired and checked by reading, not compiled, not run on hardware
//
#include <hip/hip_runtime.h>
#include <hip/hip_bf16.h>
#include <math.h>

// Problem constants (from reference)
#define BB 512
#define NN 128
#define EE 256
#define HH 8
#define DD 32
#define NEGV -1000000000.0f
#define CLIPV 10.0f
// scale = 1/sqrt(32)
#define SCALE 0.17677669529663687f

typedef __attribute__((ext_vector_type(16))) __bf16 v16bf;
typedef __attribute__((ext_vector_type(8)))  __bf16 v8bf;
typedef __attribute__((ext_vector_type(8)))  float  v8f;

// ---------------------------------------------------------------------------
// Small precompute kernels (trivial FLOPs, run once)
// ---------------------------------------------------------------------------

// Weff[k][e] = sum_j Wqkv[k][512+j] * Wmlp[e][j]   (fold Wmlp into logit_k proj)
// beff[e]   = sum_j bqkv[512+j]    * Wmlp[e][j]
// w0[k]     = sum_j Wqkv[k][512+j] * bmlp[j]
// s0        = sum_j bqkv[512+j]    * bmlp[j]
__global__ __launch_bounds__(256) void pre_kernel(
    const float* __restrict__ Wqkv, const float* __restrict__ Wmlp,
    const float* __restrict__ bqkv, const float* __restrict__ bmlp,
    float* __restrict__ Weff, float* __restrict__ beff,
    float* __restrict__ w0, float* __restrict__ s0) {
  const int blk = blockIdx.x, t = threadIdx.x;
  if (blk < 256) {
    const float* wr = Wqkv + blk * 768 + 512;
    const float* mr = Wmlp + t * 256;
    float s = 0.f;
    for (int j = 0; j < 256; ++j) s += wr[j] * mr[j];
    Weff[blk * 256 + t] = s;
  } else if (blk == 256) {
    const float* mr = Wmlp + t * 256;
    float s = 0.f;
    for (int j = 0; j < 256; ++j) s += bqkv[512 + j] * mr[j];
    beff[t] = s;
  } else {
    const float* wr = Wqkv + t * 768 + 512;
    float s = 0.f;
    for (int j = 0; j < 256; ++j) s += wr[j] * bmlp[j];
    w0[t] = s;
    if (t == 0) {
      float z = 0.f;
      for (int j = 0; j < 256; ++j) z += bqkv[512 + j] * bmlp[j];
      s0[0] = z;
    }
  }
}

// graph_emb[b][e] = mean_n X[b][n][e]
__global__ __launch_bounds__(256) void ge_kernel(const float* __restrict__ X,
                                                 float* __restrict__ ge) {
  const int b = blockIdx.x, t = threadIdx.x;
  float s = 0.f;
  for (int n = 0; n < NN; ++n) s += X[((size_t)b * NN + n) * EE + t];
  ge[b * EE + t] = s * (1.0f / NN);
}

// qbase[b][j] = ge[b]·Wfix[:,j] + bfix[j] + X[b][0]·Wstep[0:E,j] + bstep[j]
__global__ __launch_bounds__(256) void qbase_kernel(
    const float* __restrict__ X, const float* __restrict__ ge,
    const float* __restrict__ Wfix, const float* __restrict__ bfix,
    const float* __restrict__ Wstep, const float* __restrict__ bstep,
    float* __restrict__ qbase) {
  const int b = blockIdx.x, j = threadIdx.x;
  const float* g = ge + b * EE;
  const float* f0 = X + (size_t)b * NN * EE;  // first node row
  float s = bfix[j] + bstep[j];
  for (int i = 0; i < EE; ++i)
    s += g[i] * Wfix[i * EE + j] + f0[i] * Wstep[i * EE + j];
  qbase[b * EE + j] = s;
}

// c0[m] = X[m]·w0 + s0   (m over B*N rows)
__global__ __launch_bounds__(256) void c0_kernel(const float* __restrict__ X,
                                                 const float* __restrict__ w0,
                                                 const float* __restrict__ s0,
                                                 float* __restrict__ c0) {
  const size_t m = (size_t)blockIdx.x * 256 + threadIdx.x;
  const float* xr = X + m * EE;
  float s = s0[0];
  for (int k = 0; k < EE; ++k) s += xr[k] * w0[k];
  c0[m] = s;
}

// ---------------------------------------------------------------------------
// WMMA GEMM:  out(M=65536 x 256, bf16) = A(f32, 65536x256) @ Bsrc-slice + bias
// Block = 256 threads = 8 wave32s; block tile 64x32; wave tile 16x16;
// K=256 in 8 chunks of 32 via v_wmma_f32_16x16x32_bf16.
// ---------------------------------------------------------------------------
#define LDS_STRIDE 264  // bf16 elems; 528B rows -> 16B-aligned, bank-skewed

__global__ __launch_bounds__(256) void gemm_wmma(
    const float* __restrict__ A, const float* __restrict__ Bsrc,
    int k0, int n0, int ldb, const float* __restrict__ bias,
    __bf16* __restrict__ out) {
  __shared__ __bf16 As[64 * LDS_STRIDE];
  __shared__ __bf16 Bs[32 * LDS_STRIDE];
  const int t = threadIdx.x;
  const int bM = blockIdx.x;  // 1024 tiles of 64 rows
  const int bN = blockIdx.y;  // 8 tiles of 32 cols

  // Stage A tile (64x256 f32 -> bf16), coalesced over columns
  const float* Ag = A + (size_t)bM * 64 * EE;
  for (int i = 0; i < 64; ++i) {
    const int e = i * 256 + t;
    const int row = e >> 8, col = e & 255;
    As[row * LDS_STRIDE + col] = (__bf16)Ag[row * EE + col];
  }
  // Stage B tile transposed: Bs[n][k] = Bsrc[(k0+k)*ldb + n0 + bN*32 + n]
  for (int i = 0; i < 32; ++i) {
    const int e = i * 256 + t;
    const int n = e & 31, k = e >> 5;
    Bs[n * LDS_STRIDE + k] =
        (__bf16)Bsrc[(size_t)(k0 + k) * ldb + n0 + bN * 32 + n];
  }
  __syncthreads();

  const int wave = t >> 5, lane = t & 31;
  const int wM = wave & 3, wN = wave >> 2;
  const int lrow = lane & 15, lhalf = lane >> 4;

  v8f acc = {};
  const __bf16* Ap = &As[(wM * 16 + lrow) * LDS_STRIDE];
  const __bf16* Bp = &Bs[(wN * 16 + lrow) * LDS_STRIDE];
#pragma unroll
  for (int kc = 0; kc < 8; ++kc) {
    // A 16x32 bf16 fragment: elems 0..7 -> K=kc*32+lhalf*8+0..7, 8..15 -> +16
    const v8bf alo = *(const v8bf*)(Ap + kc * 32 + lhalf * 8);
    const v8bf ahi = *(const v8bf*)(Ap + kc * 32 + lhalf * 8 + 16);
    // B 32x16 bf16 fragment: col = lane&15, elems 0..7 -> K=kc*32+lhalf*8+0..7, 8..15 -> +16
    const v8bf blo = *(const v8bf*)(Bp + kc * 32 + lhalf * 8);
    const v8bf bhi = *(const v8bf*)(Bp + kc * 32 + lhalf * 8 + 16);
    const v16bf a = __builtin_shufflevector(alo, ahi, 0, 1, 2, 3, 4, 5, 6, 7,
                                            8, 9, 10, 11, 12, 13, 14, 15);
    const v16bf b = __builtin_shufflevector(blo, bhi, 0, 1, 2, 3, 4, 5, 6, 7,
                                            8, 9, 10, 11, 12, 13, 14, 15);
    acc = __builtin_amdgcn_wmma_f32_16x16x32_bf16(false, a, false, b,
                                                  (short)0, acc, false, false);
  }

  // Epilogue: C layout -> lane = N col, VGPR r holds M = r (+8 for lanes 16..31)
  const int nc = bN * 32 + wN * 16 + lrow;         // 0..255
  const float bv = bias ? bias[nc] : 0.0f;
#pragma unroll
  for (int r = 0; r < 8; ++r) {
    const int mg = bM * 64 + wM * 16 + r + 8 * lhalf;
    out[(size_t)mg * EE + nc] = (__bf16)(acc[r] + bv);
  }
}

// ---------------------------------------------------------------------------
// Decode: one block per batch element, 127 sequential steps, all operands
// (k, v, lk2) resident in LDS as bf16 (192KB of the 320KB WGP LDS).
// ---------------------------------------------------------------------------
__global__ __launch_bounds__(256) void decode_kernel(
    const __bf16* __restrict__ k_bf, const __bf16* __restrict__ v_bf,
    const __bf16* __restrict__ lk2_bf, const __bf16* __restrict__ P_bf,
    const float* __restrict__ c0, const float* __restrict__ qbase,
    float* __restrict__ out) {
  extern __shared__ __align__(16) char smem[];
  __bf16* ks = (__bf16*)smem;            // 128*256
  __bf16* vs = ks + NN * EE;             // 128*256
  __bf16* ls = vs + NN * EE;             // 128*256
  float* qb  = (float*)(ls + NN * EE);   // 256
  float* q   = qb + EE;                  // 256
  float* att = q + EE;                   // 8*128
  float* ctx = att + HH * NN;            // 256
  float* lg  = ctx + EE;                 // 128
  int* vis   = (int*)(lg + NN);          // 128
  int* cur   = vis + NN;                 // 1
  float* lps = (float*)(cur + 1);        // 1

  const int b = blockIdx.x;
  const int t = threadIdx.x;

  // Stage k/v/lk2 tiles (each 64KB bf16) with 16B vector copies
  {
    const uint4* gk = (const uint4*)(k_bf + (size_t)b * NN * EE);
    const uint4* gv = (const uint4*)(v_bf + (size_t)b * NN * EE);
    const uint4* gl = (const uint4*)(lk2_bf + (size_t)b * NN * EE);
    uint4* sk = (uint4*)ks;
    uint4* sv = (uint4*)vs;
    uint4* sl = (uint4*)ls;
    for (int i = t; i < NN * EE / 8; i += 256) {
      sk[i] = gk[i];
      sv[i] = gv[i];
      sl[i] = gl[i];
    }
  }
  qb[t] = qbase[(size_t)b * EE + t];
  if (t < NN) vis[t] = (t == 0) ? 1 : 0;
  if (t == 0) { cur[0] = 0; lps[0] = 0.0f; }
  __syncthreads();

  for (int step = 0; step < NN - 1; ++step) {
    const int c = cur[0];
    // q = qbase + P[b][current]
    q[t] = qb[t] + (float)P_bf[((size_t)b * NN + c) * EE + t];
    __syncthreads();

    // attention logits: 8 heads x 128 keys, dot over d=32, masked
    for (int it = 0; it < 4; ++it) {
      const int idx = it * 256 + t;
      const int h = idx >> 7, n = idx & 127;
      const __bf16* kr = ks + n * EE + h * DD;
      const float* qr = q + h * DD;
      float s = 0.f;
#pragma unroll
      for (int d = 0; d < DD; ++d) s += qr[d] * (float)kr[d];
      att[h * NN + n] = vis[n] ? NEGV : s * SCALE;
    }
    __syncthreads();

    // softmax per head: wave w owns head w (wave32, 4 values/lane)
    {
      const int h = t >> 5, l = t & 31;
      float v4[4], m = -1e30f;
#pragma unroll
      for (int i = 0; i < 4; ++i) {
        v4[i] = att[h * NN + l + 32 * i];
        m = fmaxf(m, v4[i]);
      }
      for (int off = 16; off > 0; off >>= 1)
        m = fmaxf(m, __shfl_xor(m, off, 32));
      float s = 0.f;
#pragma unroll
      for (int i = 0; i < 4; ++i) { v4[i] = __expf(v4[i] - m); s += v4[i]; }
      for (int off = 16; off > 0; off >>= 1) s += __shfl_xor(s, off, 32);
      const float inv = 1.0f / s;
#pragma unroll
      for (int i = 0; i < 4; ++i) att[h * NN + l + 32 * i] = v4[i] * inv;
    }
    __syncthreads();

    // ctx[e] (e = h*32+d): weighted sum over keys
    {
      const int h = t >> 5;
      float s = 0.f;
      for (int n = 0; n < NN; ++n)
        s += att[h * NN + n] * (float)vs[n * EE + t];
      ctx[t] = s;
    }
    __syncthreads();

    // logits[n] = tanh(scale*(ctx·lk2[n] + c0[n])) * 10, masked
    if (t < NN) {
      const __bf16* lr = ls + t * EE;
      float s = c0[(size_t)b * NN + t];
      for (int e = 0; e < EE; ++e) s += ctx[e] * (float)lr[e];
      const float x = tanhf(s * SCALE) * CLIPV;
      lg[t] = vis[t] ? NEGV : x;
    }
    __syncthreads();

    // wave 0: argmax (first-index tie break) + log_softmax chosen value
    if (t < 32) {
      float m = -1e30f;
      int am = 0;
#pragma unroll
      for (int i = 0; i < 4; ++i) {
        const int n = t + 32 * i;
        const float vv = lg[n];
        if (vv > m || (vv == m && n < am)) { m = vv; am = n; }
      }
      for (int off = 16; off > 0; off >>= 1) {
        const float om = __shfl_xor(m, off, 32);
        const int oam = __shfl_xor(am, off, 32);
        if (om > m || (om == m && oam < am)) { m = om; am = oam; }
      }
      float s = 0.f;
#pragma unroll
      for (int i = 0; i < 4; ++i) s += __expf(lg[t + 32 * i] - m);
      for (int off = 16; off > 0; off >>= 1) s += __shfl_xor(s, off, 32);
      if (t == 0) {
        lps[0] += lg[am] - m - __logf(s);  // log_softmax at argmax
        vis[am] = 1;
        cur[0] = am;
      }
    }
    __syncthreads();
  }
  if (t == 0) out[b] = lps[0];
}

// ---------------------------------------------------------------------------
// Launch
// ---------------------------------------------------------------------------
extern "C" void kernel_launch(void* const* d_in, const int* in_sizes, int n_in,
                              void* d_out, int out_size, void* d_ws,
                              size_t ws_size, hipStream_t stream) {
  (void)in_sizes; (void)n_in; (void)out_size; (void)ws_size;
  const float* X     = (const float*)d_in[0];
  const float* Wqkv  = (const float*)d_in[1];
  const float* bqkv  = (const float*)d_in[2];
  const float* Wfix  = (const float*)d_in[3];
  const float* bfix  = (const float*)d_in[4];
  const float* Wstep = (const float*)d_in[5];
  const float* bstep = (const float*)d_in[6];
  const float* Wmlp  = (const float*)d_in[7];
  const float* bmlp  = (const float*)d_in[8];
  float* out = (float*)d_out;

  // Workspace layout
  char* ws = (char*)d_ws;
  const size_t MB32 = (size_t)BB * NN * EE * sizeof(__bf16);  // 32MB each
  __bf16* k_bf   = (__bf16*)(ws);
  __bf16* v_bf   = (__bf16*)(ws + MB32);
  __bf16* lk2_bf = (__bf16*)(ws + 2 * MB32);
  __bf16* P_bf   = (__bf16*)(ws + 3 * MB32);
  char* p = ws + 4 * MB32;
  float* c0    = (float*)p;            p += (size_t)BB * NN * sizeof(float);
  float* qbase = (float*)p;            p += (size_t)BB * EE * sizeof(float);
  float* ge    = (float*)p;            p += (size_t)BB * EE * sizeof(float);
  float* Weff  = (float*)p;            p += (size_t)EE * EE * sizeof(float);
  float* beff  = (float*)p;            p += (size_t)EE * sizeof(float);
  float* w0    = (float*)p;            p += (size_t)EE * sizeof(float);
  float* s0    = (float*)p;

  // Phase 0: fold Wmlp/bmlp into logit path; graph embedding; q base
  pre_kernel<<<dim3(258), 256, 0, stream>>>(Wqkv, Wmlp, bqkv, bmlp, Weff, beff,
                                            w0, s0);
  ge_kernel<<<dim3(BB), 256, 0, stream>>>(X, ge);
  qbase_kernel<<<dim3(BB), 256, 0, stream>>>(X, ge, Wfix, bfix, Wstep, bstep,
                                             qbase);
  c0_kernel<<<dim3(BB * NN / 256), 256, 0, stream>>>(X, w0, s0, c0);

  // Phase 1: bulk WMMA GEMMs over M = B*N = 65536 rows, K = 256
  const dim3 ggrid(BB * NN / 64, EE / 32);
  gemm_wmma<<<ggrid, 256, 0, stream>>>(X, Wqkv, 0, 0, 3 * EE, bqkv, k_bf);
  gemm_wmma<<<ggrid, 256, 0, stream>>>(X, Wqkv, 0, EE, 3 * EE, bqkv + EE, v_bf);
  gemm_wmma<<<ggrid, 256, 0, stream>>>(X, Weff, 0, 0, EE, beff, lk2_bf);
  gemm_wmma<<<ggrid, 256, 0, stream>>>(X, Wstep, EE, 0, EE, nullptr, P_bf);

  // Phase 2: LDS-resident sequential decode, one block per batch element
  const size_t smem = (size_t)3 * NN * EE * sizeof(__bf16)            // k,v,lk2
                    + (size_t)(EE + EE + HH * NN + EE + NN) * sizeof(float)
                    + (size_t)(NN + 1) * sizeof(int) + sizeof(float) + 64;
  decode_kernel<<<dim3(BB), 256, smem, stream>>>(k_bf, v_bf, lk2_bf, P_bf, c0,
                                                 qbase, out);
}
